// GCN_52871047413869
// MI455X (gfx1250) — matrix operand order repaired and checked by reading, not compiled, not run on hardware
//
#include <hip/hip_runtime.h>
#include <hip/hip_bf16.h>

typedef __attribute__((ext_vector_type(16))) _Float16 v16h;
typedef __attribute__((ext_vector_type(8)))  float    v8f;

#define N_NODES 100000
#define N_EDGES 1600000
#define N_GRAPHS 1000

// ---------------------------------------------------------------------------
// deg[d] += 1 for every edge dst (in-degree); self-loop +1 folded in dinv pass
// ---------------------------------------------------------------------------
__global__ void deg_kernel(const long long* __restrict__ dst, float* __restrict__ deg, int E) {
    int e = blockIdx.x * blockDim.x + threadIdx.x;
    if (e >= E) return;
    atomicAdd(&deg[dst[e]], 1.0f);
}

__global__ void dinv_kernel(float* __restrict__ dinv, int n) {
    int i = blockIdx.x * blockDim.x + threadIdx.x;
    if (i >= n) return;
    float deg = dinv[i] + 1.0f;              // +1 self loop; always > 0
    dinv[i] = __frsqrt_rn(deg);
}

// ---------------------------------------------------------------------------
// Layer-1 linear: h1 = x @ W1   (K=6 -> plain VALU, W1 staged in LDS)
// ---------------------------------------------------------------------------
__global__ void gemm1_kernel(const float* __restrict__ x, const float* __restrict__ W1,
                             float* __restrict__ h1, int n) {
    __shared__ float Ws[6 * 64];
    for (int i = threadIdx.x; i < 6 * 64; i += blockDim.x) Ws[i] = W1[i];
    __syncthreads();
    int idx  = blockIdx.x * blockDim.x + threadIdx.x;
    int node = idx >> 6;
    int f    = idx & 63;
    if (node >= n) return;
    const float* xr = x + node * 6;
    float acc = 0.0f;
#pragma unroll
    for (int k = 0; k < 6; ++k) acc = fmaf(xr[k], Ws[k * 64 + f], acc);
    h1[idx] = acc;
}

// ---------------------------------------------------------------------------
// Edge aggregation: out[d] += h[s] * dinv[s]*dinv[d].  One wave32 per edge,
// lane f = coalesced 128B gather per instruction, fp32 global atomics.
// ---------------------------------------------------------------------------
__global__ void edge_agg_kernel(const long long* __restrict__ src,
                                const long long* __restrict__ dst,
                                const float* __restrict__ dinv,
                                const float* __restrict__ h,
                                float* __restrict__ out, int E, int F) {
    int e = blockIdx.x * (blockDim.x >> 5) + (threadIdx.x >> 5);
    if (e >= E) return;
    int lane = threadIdx.x & 31;
    long long s = src[e];
    long long d = dst[e];
    float norm = dinv[s] * dinv[d];
    const float* hs = h + s * (long long)F;
    float*       od = out + d * (long long)F;
    for (int f = lane; f < F; f += 32)
        atomicAdd(&od[f], hs[f] * norm);
}

// ---------------------------------------------------------------------------
// Self-loop + bias + ReLU (in place on aggregate buffer, no atomics):
// g = relu(g + h*dinv^2 + b)
// ---------------------------------------------------------------------------
__global__ void sbr_kernel(const float* __restrict__ h, const float* __restrict__ dinv,
                           const float* __restrict__ b, float* __restrict__ g,
                           int n, int logF) {
    long long idx = (long long)blockIdx.x * blockDim.x + threadIdx.x;
    int F = 1 << logF;
    if (idx >= (long long)n << logF) return;
    int node = (int)(idx >> logF);
    int f    = (int)(idx & (F - 1));
    float di = dinv[node];
    float v  = g[idx] + h[idx] * di * di + b[f];
    g[idx] = v > 0.0f ? v : 0.0f;
}

// ---------------------------------------------------------------------------
// Layer-2 linear via WMMA: h2[M,128] = g1[M,64] @ W2[64,128]
// M = 100000 = 6250 exact 16-row tiles. One wave per 16-row strip; A fragment
// reused across 8 column tiles; K = 64 -> 2 x v_wmma_f32_16x16x32_f16 / tile.
//
// W2 is pre-packed into LDS in per-lane fragment order so each B fragment is
// a single contiguous 32-byte LDS load per lane (2x ds_load_b128) instead of
// 32 strided ds_load_u16 + v_mov_b16 packing.
//
// Fragment layouts (ISA 7.12.2, f16):
//   A 16x32: lane l: row=l&15, hi=l>>4; elem j<8: K=hi*8+j; j>=8: K=16+hi*8+j-8
//   B 32x16: lane l: col=l&15;          elem j:  K as in A (j<8: hi*8+j; j>=8: 16+hi*8+j-8)
//   D      : elem i: M=i+8*(l>>4), N=l&15
// ---------------------------------------------------------------------------
__global__ void gemm2_wmma_kernel(const float* __restrict__ A,   // [M,64]
                                  const float* __restrict__ W2,  // [64,128]
                                  float* __restrict__ C,         // [M,128]
                                  int mtiles) {
    // Wpack[kt][ct][lane][j] : 2*8*32*16 halves = 16 KB
    __shared__ alignas(32) _Float16 Wpack[2 * 8 * 32 * 16];

    // one-time pack: 512 fragments, each thread packs 2 of them
    for (int frag = threadIdx.x; frag < 2 * 8 * 32; frag += blockDim.x) {
        int lane = frag & 31;
        int ct   = (frag >> 5) & 7;
        int kt   = frag >> 8;
        int hi   = lane >> 4;
        int colN = ct * 16 + (lane & 15);
        _Float16*    dp = &Wpack[frag * 16];
        const float* sp = W2 + (kt * 32 + hi * 8) * 128 + colN;
#pragma unroll
        for (int j = 0; j < 16; ++j) dp[j] = (_Float16)sp[(j + (j & 8)) * 128];
    }
    __syncthreads();

    const int wave  = threadIdx.x >> 5;
    const int lane  = threadIdx.x & 31;
    const int mtile = blockIdx.x * (blockDim.x >> 5) + wave;
    if (mtile >= mtiles) return;                 // wave-uniform exit

    const int row = mtile * 16 + (lane & 15);
    const int hi  = lane >> 4;                   // 0 or 1
    const int col = lane & 15;

    // A fragments: 8 x float4 vector loads, convert to f16
    const float* ar = A + (long long)row * 64 + hi * 8;
    float4 q0 = *(const float4*)(ar +  0);   // K = hi*8 + 0..3
    float4 q1 = *(const float4*)(ar +  4);   // K = hi*8 + 4..7
    float4 q2 = *(const float4*)(ar + 16);   // K = 16+hi*8 + 0..3
    float4 q3 = *(const float4*)(ar + 20);   // K = 16+hi*8 + 4..7
    float4 q4 = *(const float4*)(ar + 32);   // K = 32+hi*8 + 0..3
    float4 q5 = *(const float4*)(ar + 36);   // K = 32+hi*8 + 4..7
    float4 q6 = *(const float4*)(ar + 48);   // K = 48+hi*8 + 0..3
    float4 q7 = *(const float4*)(ar + 52);   // K = 48+hi*8 + 4..7

    v16h a0, a1;
    a0[0]  = (_Float16)q0.x; a0[1]  = (_Float16)q0.y; a0[2]  = (_Float16)q0.z; a0[3]  = (_Float16)q0.w;
    a0[4]  = (_Float16)q1.x; a0[5]  = (_Float16)q1.y; a0[6]  = (_Float16)q1.z; a0[7]  = (_Float16)q1.w;
    a0[8]  = (_Float16)q2.x; a0[9]  = (_Float16)q2.y; a0[10] = (_Float16)q2.z; a0[11] = (_Float16)q2.w;
    a0[12] = (_Float16)q3.x; a0[13] = (_Float16)q3.y; a0[14] = (_Float16)q3.z; a0[15] = (_Float16)q3.w;
    a1[0]  = (_Float16)q4.x; a1[1]  = (_Float16)q4.y; a1[2]  = (_Float16)q4.z; a1[3]  = (_Float16)q4.w;
    a1[4]  = (_Float16)q5.x; a1[5]  = (_Float16)q5.y; a1[6]  = (_Float16)q5.z; a1[7]  = (_Float16)q5.w;
    a1[8]  = (_Float16)q6.x; a1[9]  = (_Float16)q6.y; a1[10] = (_Float16)q6.z; a1[11] = (_Float16)q6.w;
    a1[12] = (_Float16)q7.x; a1[13] = (_Float16)q7.y; a1[14] = (_Float16)q7.z; a1[15] = (_Float16)q7.w;

    const v16h* wp = (const v16h*)Wpack;     // fragment index = kt*256 + ct*32 + lane

    v8f acc[8];
#pragma unroll
    for (int ct = 0; ct < 8; ++ct) acc[ct] = (v8f){0.f, 0.f, 0.f, 0.f, 0.f, 0.f, 0.f, 0.f};

#pragma unroll
    for (int ct = 0; ct < 8; ++ct) {
        v16h b0 = wp[ct * 32 + lane];          // kt=0 fragment, contiguous 32B
        v16h b1 = wp[256 + ct * 32 + lane];    // kt=1 fragment
        acc[ct] = __builtin_amdgcn_wmma_f32_16x16x32_f16(
            false, a0, false, b0, (short)0, acc[ct], false, false);
        acc[ct] = __builtin_amdgcn_wmma_f32_16x16x32_f16(
            false, a1, false, b1, (short)0, acc[ct], false, false);
    }

    // store D: elem i -> (M = mtile*16 + i + 8*hi, N = ct*16 + col)
#pragma unroll
    for (int ct = 0; ct < 8; ++ct) {
        float* cr = C + ((long long)mtile * 16 + 8 * hi) * 128 + ct * 16 + col;
#pragma unroll
        for (int i = 0; i < 8; ++i)
            cr[(long long)i * 128] = acc[ct][i];
    }
}

// ---------------------------------------------------------------------------
// Mean-pool accumulation: wave per node, atomic add into pooled[batch[node]]
// ---------------------------------------------------------------------------
__global__ void pool_kernel(const long long* __restrict__ batch,
                            const float* __restrict__ g2,
                            float* __restrict__ pooled, float* __restrict__ counts, int n) {
    int node = blockIdx.x * (blockDim.x >> 5) + (threadIdx.x >> 5);
    if (node >= n) return;
    int lane = threadIdx.x & 31;
    long long gidx = batch[node];
    const float* r = g2 + (long long)node * 128;
    float*       p = pooled + gidx * 128;
#pragma unroll
    for (int j = 0; j < 4; ++j)
        atomicAdd(&p[lane + 32 * j], r[lane + 32 * j]);
    if (lane == 0) atomicAdd(&counts[gidx], 1.0f);
}

// ---------------------------------------------------------------------------
// Final projection: out[g, c] = (pooled[g]/max(count,1)) @ Wfc + bfc
// ---------------------------------------------------------------------------
__global__ void final_kernel(const float* __restrict__ pooled, const float* __restrict__ counts,
                             const float* __restrict__ Wfc, const float* __restrict__ bfc,
                             float* __restrict__ out, int G) {
    int idx = blockIdx.x * blockDim.x + threadIdx.x;
    if (idx >= G * 2) return;
    int g = idx >> 1;
    int c = idx & 1;
    float inv = 1.0f / fmaxf(counts[g], 1.0f);
    float acc = bfc[c];
    const float* p = pooled + (long long)g * 128;
#pragma unroll 4
    for (int f = 0; f < 128; ++f)
        acc = fmaf(p[f] * inv, Wfc[f * 2 + c], acc);
    out[idx] = acc;
}

// ---------------------------------------------------------------------------
extern "C" void kernel_launch(void* const* d_in, const int* in_sizes, int n_in,
                              void* d_out, int out_size, void* d_ws, size_t ws_size,
                              hipStream_t stream) {
    const float*     x    = (const float*)d_in[0];
    const long long* ei   = (const long long*)d_in[1];   // [2, E] int64
    const long long* bat  = (const long long*)d_in[2];   // [N] int64
    const float*     W1   = (const float*)d_in[3];
    const float*     b1   = (const float*)d_in[4];
    const float*     W2   = (const float*)d_in[5];
    const float*     b2   = (const float*)d_in[6];
    const float*     Wfc  = (const float*)d_in[7];
    const float*     bfc  = (const float*)d_in[8];
    float* out = (float*)d_out;

    const long long* srcp = ei;
    const long long* dstp = ei + N_EDGES;

    // ---- workspace layout (running offset, 256B aligned) ----
    char* ws = (char*)d_ws;
    size_t off = 0;
    auto alloc = [&](size_t bytes) -> char* {
        char* p = ws + off;
        off += (bytes + 255) & ~(size_t)255;
        return p;
    };
    float* dinv   = (float*)alloc((size_t)N_NODES * 4);           // deg -> dinv
    float* h1     = (float*)alloc((size_t)N_NODES * 64 * 4);      // x @ W1
    float* g1     = (float*)alloc((size_t)N_NODES * 64 * 4);      // aggregated layer1
    float* h2     = (float*)alloc((size_t)N_NODES * 128 * 4);     // g1 @ W2 (WMMA)
    float* g2     = (float*)alloc((size_t)N_NODES * 128 * 4);     // aggregated layer2
    float* pooled = (float*)alloc((size_t)N_GRAPHS * 128 * 4);
    float* counts = (float*)alloc((size_t)N_GRAPHS * 4);
    (void)ws_size;

    // ---- zero accumulation buffers (graph-capture-safe async memsets) ----
    hipMemsetAsync(dinv,   0, (size_t)N_NODES * 4, stream);
    hipMemsetAsync(g1,     0, (size_t)N_NODES * 64 * 4, stream);
    hipMemsetAsync(g2,     0, (size_t)N_NODES * 128 * 4, stream);
    hipMemsetAsync(pooled, 0, (size_t)N_GRAPHS * 128 * 4, stream);
    hipMemsetAsync(counts, 0, (size_t)N_GRAPHS * 4, stream);

    // ---- degree + symmetric norm ----
    deg_kernel<<<(N_EDGES + 255) / 256, 256, 0, stream>>>(dstp, dinv, N_EDGES);
    dinv_kernel<<<(N_NODES + 255) / 256, 256, 0, stream>>>(dinv, N_NODES);

    // ---- layer 1 ----
    gemm1_kernel<<<(N_NODES * 64 + 255) / 256, 256, 0, stream>>>(x, W1, h1, N_NODES);
    edge_agg_kernel<<<(N_EDGES + 7) / 8, 256, 0, stream>>>(srcp, dstp, dinv, h1, g1, N_EDGES, 64);
    sbr_kernel<<<(N_NODES * 64 + 255) / 256, 256, 0, stream>>>(h1, dinv, b1, g1, N_NODES, 6);

    // ---- layer 2 (WMMA GEMM) ----
    const int mtiles = N_NODES / 16;        // 6250, exact
    gemm2_wmma_kernel<<<(mtiles + 7) / 8, 256, 0, stream>>>(g1, W2, h2, mtiles);
    edge_agg_kernel<<<(N_EDGES + 7) / 8, 256, 0, stream>>>(srcp, dstp, dinv, h2, g2, N_EDGES, 128);
    sbr_kernel<<<((long long)N_NODES * 128 + 255) / 256, 256, 0, stream>>>(h2, dinv, b2, g2, N_NODES, 7);

    // ---- mean pool + classifier ----
    pool_kernel<<<(N_NODES + 7) / 8, 256, 0, stream>>>(bat, g2, pooled, counts, N_NODES);
    final_kernel<<<(N_GRAPHS * 2 + 255) / 256, 256, 0, stream>>>(pooled, counts, Wfc, bfc, out, N_GRAPHS);
}
